// MultiHeadAttentionWithRelativePositionBias_25254407700719
// MI455X (gfx1250) — matrix operand order repaired and checked by reading, not compiled, not run on hardware
//
#include <hip/hip_runtime.h>

// ---------------------------------------------------------------------------
// MHA with relative position bias, CDNA5 (gfx1250) WMMA bf16 implementation.
// B=2, S=2048, D_MODEL=768, H=12, DEPTH=64
// ---------------------------------------------------------------------------

#define S_LEN  2048
#define DMODEL 768
#define NHEAD  12
#define DHEAD  64
#define NBATCH 2
#define NEGVAL (-1.0e9f)

typedef __attribute__((ext_vector_type(16))) __bf16 v16bf;
typedef __attribute__((ext_vector_type(8)))  __bf16 v8bf;
typedef __attribute__((ext_vector_type(8)))  float  v8f;
typedef int v4i_vs __attribute__((vector_size(16)));  // matches builtin param

#define AS1 __attribute__((address_space(1)))
#define AS3 __attribute__((address_space(3)))

#if defined(__has_builtin)
#if __has_builtin(__builtin_amdgcn_global_load_async_to_lds_b128)
#define HAVE_ASYNC_LDS 1
#endif
#endif
#ifndef HAVE_ASYNC_LDS
#define HAVE_ASYNC_LDS 0
#endif

// Copy 16 bytes (8 x bf16) global -> LDS on the async path (ASYNCcnt) when
// the toolchain exposes the gfx1250 builtin; else synchronous fallback.
__device__ __forceinline__ void async_copy_b128(const __bf16* g, __bf16* l) {
#if HAVE_ASYNC_LDS
  __builtin_amdgcn_global_load_async_to_lds_b128(
      (AS1 v4i_vs*)(void*)g, (AS3 v4i_vs*)(void*)l, 0, 0);
#else
  *(v8bf*)l = *(const v8bf*)g;
#endif
}

__device__ __forceinline__ void wait_async() {
#if HAVE_ASYNC_LDS
#if __has_builtin(__builtin_amdgcn_s_wait_asynccnt)
  __builtin_amdgcn_s_wait_asynccnt(0);
#else
  asm volatile("s_wait_asynccnt 0x0" ::: "memory");
#endif
#endif
}

__device__ __forceinline__ v8f vzero8() {
  v8f z;
#pragma unroll
  for (int i = 0; i < 8; ++i) z[i] = 0.0f;
  return z;
}

__device__ __forceinline__ v8f wmma_bf16(v16bf a, v16bf b, v8f c) {
  // v_wmma_f32_16x16x32_bf16, f32 accumulate
  return __builtin_amdgcn_wmma_f32_16x16x32_bf16(
      /*neg_a=*/false, a, /*neg_b=*/false, b,
      /*c_mod=*/(short)0, c, /*reuse_a=*/false, /*reuse_b=*/false);
}

// A-fragment (16xK slice, 16-bit): lane<16 holds row=lane, K
// [koff..koff+15] contiguous; lane>=16 gets +16.
__device__ __forceinline__ v16bf load_afrag(const __bf16* base, int stride,
                                            int lane, int koff) {
  int row = lane & 15;
  int k0  = koff + ((lane >> 4) << 4);
  const __bf16* p = base + row * stride + k0;
  union { v16bf v; v8bf h[2]; } u;
  u.h[0] = *(const v8bf*)(p);
  u.h[1] = *(const v8bf*)(p + 8);
  return u.v;
}

// B-fragment (32x16 slice, 16-bit): LDS stored as [N][K] row-major so each
// lane reads 16 contiguous K values: col n = ntile*16 + lane%16,
// k = kk + (lane/16)*16 .. +15.
__device__ __forceinline__ v16bf load_bfrag(const __bf16* base, int stride,
                                            int lane, int ntile, int kk) {
  int n  = (ntile << 4) + (lane & 15);
  int k0 = kk + ((lane >> 4) << 4);
  return *(const v16bf*)(base + n * stride + k0);
}

// ---------------------------------------------------------------------------
// Phase 1: X[4096x768] f32 @ W[768x768] f32 + bias -> bf16 head-split
// out[b][h][s][d].  Grid (32, 12), block 256 (8 waves).
// Workgroup tile M=128 N=64, K-step 64 (8 WMMA per barrier).
// ---------------------------------------------------------------------------
__global__ __launch_bounds__(256) void proj_kernel(
    const float* __restrict__ X, const float* __restrict__ W,
    const float* __restrict__ bias, __bf16* __restrict__ out) {
  __shared__ __bf16 ldsA[128 * 64];  // [m][k]
  __shared__ __bf16 ldsB[64 * 64];   // [n][k]  (W transposed)

  const int tid  = threadIdx.x;
  const int lane = tid & 31;
  const int wave = tid >> 5;
  const int m0   = blockIdx.x * 128;
  const int n0   = blockIdx.y * 64;  // head boundary (DHEAD==64)

  v8f acc[4];
#pragma unroll
  for (int nt = 0; nt < 4; ++nt) acc[nt] = vzero8();

  for (int k0 = 0; k0 < DMODEL; k0 += 64) {
    __syncthreads();
    {  // A tile: 128x64 f32 -> bf16, 32 elems per thread
      int row = tid >> 1;
      int kh  = (tid & 1) * 32;
      const float* src = X + (size_t)(m0 + row) * DMODEL + k0 + kh;
#pragma unroll
      for (int half = 0; half < 2; ++half) {
        v16bf tmp;
#pragma unroll
        for (int i = 0; i < 16; ++i) tmp[i] = (__bf16)src[half * 16 + i];
        *(v16bf*)&ldsA[row * 64 + kh + half * 16] = tmp;
      }
    }
    {  // B tile: W[k0+k][n0+n] -> ldsB[n][k], 16 elems per thread
      int k   = tid & 63;
      int n16 = (tid >> 6) * 16;
      const float* src = W + (size_t)(k0 + k) * DMODEL + n0 + n16;
#pragma unroll
      for (int i = 0; i < 16; ++i) ldsB[(n16 + i) * 64 + k] = (__bf16)src[i];
    }
    __syncthreads();

    v16bf a0 = load_afrag(&ldsA[wave * 16 * 64], 64, lane, 0);
    v16bf a1 = load_afrag(&ldsA[wave * 16 * 64], 64, lane, 32);
#pragma unroll
    for (int nt = 0; nt < 4; ++nt) {
      acc[nt] = wmma_bf16(a0, load_bfrag(ldsB, 64, lane, nt, 0), acc[nt]);
      acc[nt] = wmma_bf16(a1, load_bfrag(ldsB, 64, lane, nt, 32), acc[nt]);
    }
  }

  const int h = blockIdx.y;  // one head per N-block
#pragma unroll
  for (int nt = 0; nt < 4; ++nt) {
    int d = nt * 16 + (lane & 15);
    float bv = bias[n0 + d];
#pragma unroll
    for (int r = 0; r < 8; ++r) {
      int row = m0 + wave * 16 + r + ((lane >> 4) << 3);  // [0,4096)
      int s   = row & (S_LEN - 1);
      int b   = row >> 11;
      out[((((size_t)b * NHEAD + h) * S_LEN) + s) * DHEAD + d] =
          (__bf16)(acc[nt][r] + bv);
    }
  }
}

// ---------------------------------------------------------------------------
// Phase 2: fused attention (flash-style online softmax).
// Grid (S/64, B*H), block 128 (4 waves); each wave owns 16 queries.
// Q/K tiles staged via async global->LDS (ASYNCcnt); V transposed via VGPRs.
// ---------------------------------------------------------------------------
__global__ __launch_bounds__(128) void attn_kernel(
    const __bf16* __restrict__ Qh, const __bf16* __restrict__ Kh,
    const __bf16* __restrict__ Vh, const int* __restrict__ posArr,
    const float* __restrict__ mask, const float* __restrict__ relBias,
    __bf16* __restrict__ Oout) {
  __shared__ __bf16 ldsQ[64 * 64];     // [q][d]
  __shared__ __bf16 ldsK[64 * 64];     // [key][d]   (== B-layout for QK^T)
  __shared__ __bf16 ldsVt[64 * 64];    // [d][key]   (== B-layout for P*V)
  __shared__ __bf16 ldsP[4][16 * 64];  // per-wave P tile [q][key]

  const int tid  = threadIdx.x;
  const int lane = tid & 31;
  const int wave = tid >> 5;
  const int bh = blockIdx.y;
  const int b  = bh / NHEAD;
  const int h  = bh % NHEAD;
  const int q0 = blockIdx.x * 64;

  const size_t headOff = (size_t)bh * S_LEN * DHEAD;
  const __bf16* Qp = Qh + headOff;
  const __bf16* Kp = Kh + headOff;
  const __bf16* Vp = Vh + headOff;

  {  // stage Q tile (64x64) via async copies: 4 x b128 per thread
    int row  = tid >> 1;
    int half = (tid & 1) * 32;
    const __bf16* src = Qp + (size_t)(q0 + row) * DHEAD + half;
    __bf16* dst = &ldsQ[row * 64 + half];
#pragma unroll
    for (int c = 0; c < 4; ++c) async_copy_b128(src + c * 8, dst + c * 8);
  }
  wait_async();
  __syncthreads();

  v16bf aq[2];
  aq[0] = load_afrag(&ldsQ[wave * 16 * 64], 64, lane, 0);
  aq[1] = load_afrag(&ldsQ[wave * 16 * 64], 64, lane, 32);

  const int rowbase = q0 + wave * 16 + ((lane >> 4) << 3);  // + r -> query
  int   pi[8];
  float l[8], mrow[8];
#pragma unroll
  for (int r = 0; r < 8; ++r) {
    pi[r]   = posArr[b * S_LEN + rowbase + r];
    l[r]    = 0.0f;
    mrow[r] = -3.0e38f;
  }
  v8f Oacc[4];
#pragma unroll
  for (int nt = 0; nt < 4; ++nt) Oacc[nt] = vzero8();

  for (int kt = 0; kt < S_LEN / 64; ++kt) {
    const int kb = kt * 64;
    __syncthreads();  // previous iteration done reading ldsK/ldsVt
    {
      int row  = tid >> 1;
      int half = (tid & 1) * 32;
      // K tile: verbatim copy -> async global->LDS path
      const __bf16* ksrc = Kp + (size_t)(kb + row) * DHEAD + half;
      __bf16* kdst = &ldsK[row * 64 + half];
#pragma unroll
      for (int c = 0; c < 4; ++c) async_copy_b128(ksrc + c * 8, kdst + c * 8);
      // V tile: transpose into [d][key] through VGPRs
      union { v16bf v; __bf16 e[16]; } t0, t1;
      t0.v = *(const v16bf*)(Vp + (size_t)(kb + row) * DHEAD + half);
      t1.v = *(const v16bf*)(Vp + (size_t)(kb + row) * DHEAD + half + 16);
#pragma unroll
      for (int i = 0; i < 16; ++i) ldsVt[(half + i) * 64 + row] = t0.e[i];
#pragma unroll
      for (int i = 0; i < 16; ++i) ldsVt[(half + 16 + i) * 64 + row] = t1.e[i];
      // prefetch next tile's K/V rows into cache while we compute
      if (kt + 1 < S_LEN / 64) {
        __builtin_prefetch(Kp + (size_t)(kb + 64 + row) * DHEAD + half, 0, 0);
        __builtin_prefetch(Vp + (size_t)(kb + 64 + row) * DHEAD + half, 0, 0);
      }
    }
    wait_async();
    __syncthreads();

    // ---- logits = Q K^T  (16 queries x 64 keys per wave) ----
    v8f lg[4];
#pragma unroll
    for (int nt = 0; nt < 4; ++nt) {
      v8f c = vzero8();
      c = wmma_bf16(aq[0], load_bfrag(ldsK, 64, lane, nt, 0), c);
      c = wmma_bf16(aq[1], load_bfrag(ldsK, 64, lane, nt, 32), c);
      lg[nt] = c;
    }

    // ---- relative position bias + mask ----
    int jcol[4], pj[4];
#pragma unroll
    for (int nt = 0; nt < 4; ++nt) {
      jcol[nt] = kb + nt * 16 + (lane & 15);
      pj[nt]   = posArr[b * S_LEN + jcol[nt]];
    }
#pragma unroll
    for (int nt = 0; nt < 4; ++nt) {
#pragma unroll
      for (int r = 0; r < 8; ++r) {
        int i   = rowbase + r;
        int idx = pj[nt] - pi[r] + (S_LEN - 1);
        idx = idx < 0 ? 0 : (idx > 2 * S_LEN - 2 ? 2 * S_LEN - 2 : idx);
        float pb = relBias[(size_t)idx * NHEAD + h];
        float mv = mask[((size_t)b * S_LEN + i) * S_LEN + jcol[nt]];
        float lv = lg[nt][r] + pb;
        lg[nt][r] = lv * mv + (1.0f - mv) * NEGVAL;
      }
    }

    // ---- online softmax (rows split across 16-lane halves, wave32) ----
#pragma unroll
    for (int r = 0; r < 8; ++r) {
      float rmax = fmaxf(fmaxf(lg[0][r], lg[1][r]), fmaxf(lg[2][r], lg[3][r]));
#pragma unroll
      for (int x = 1; x < 16; x <<= 1)
        rmax = fmaxf(rmax, __shfl_xor(rmax, x, 32));
      float mnew  = fmaxf(mrow[r], rmax);
      float scale = __expf(mrow[r] - mnew);
      float rsum  = 0.0f;
#pragma unroll
      for (int nt = 0; nt < 4; ++nt) {
        float p = __expf(lg[nt][r] - mnew);
        lg[nt][r] = p;
        rsum += p;
      }
#pragma unroll
      for (int x = 1; x < 16; x <<= 1) rsum += __shfl_xor(rsum, x, 32);
      l[r]    = l[r] * scale + rsum;
      mrow[r] = mnew;
#pragma unroll
      for (int nt = 0; nt < 4; ++nt) Oacc[nt][r] *= scale;
    }

    // ---- P (C-layout f32) -> bf16 A-layout via per-wave LDS region ----
    __bf16* myP = &ldsP[wave][0];  // [16][64]
#pragma unroll
    for (int nt = 0; nt < 4; ++nt) {
#pragma unroll
      for (int r = 0; r < 8; ++r)
        myP[(r + ((lane >> 4) << 3)) * 64 + nt * 16 + (lane & 15)] =
            (__bf16)lg[nt][r];
    }
    // DS ops are in-order within a wave; fence the compiler + wait stores.
    asm volatile("s_wait_dscnt 0x0" ::: "memory");

    // ---- O += P * V ----
#pragma unroll
    for (int ks = 0; ks < 2; ++ks) {
      v16bf ap = load_afrag(myP, 64, lane, ks * 32);
#pragma unroll
      for (int nt = 0; nt < 4; ++nt) {
        v16bf bv = load_bfrag(ldsVt, 64, lane, nt, ks * 32);
        Oacc[nt] = wmma_bf16(ap, bv, Oacc[nt]);
      }
    }
  }

  // ---- normalize and write merged-head bf16 output [B][S][H*64] ----
#pragma unroll
  for (int nt = 0; nt < 4; ++nt) {
    int d = nt * 16 + (lane & 15);
#pragma unroll
    for (int r = 0; r < 8; ++r) {
      int i = rowbase + r;
      float o = Oacc[nt][r] / l[r];
      Oout[((size_t)b * S_LEN + i) * DMODEL + h * DHEAD + d] = (__bf16)o;
    }
  }
}

// ---------------------------------------------------------------------------
// Phase 3: attn_out[4096x768] bf16 @ Wo f32 + bo -> f32 d_out.
// Grid (32, 12), block 256; K-step 64; A tile staged via async global->LDS.
// ---------------------------------------------------------------------------
__global__ __launch_bounds__(256) void oproj_kernel(
    const __bf16* __restrict__ A, const float* __restrict__ W,
    const float* __restrict__ bias, float* __restrict__ Y) {
  __shared__ __bf16 ldsA[128 * 64];
  __shared__ __bf16 ldsB[64 * 64];

  const int tid  = threadIdx.x;
  const int lane = tid & 31;
  const int wave = tid >> 5;
  const int m0   = blockIdx.x * 128;
  const int n0   = blockIdx.y * 64;

  v8f acc[4];
#pragma unroll
  for (int nt = 0; nt < 4; ++nt) acc[nt] = vzero8();

  for (int k0 = 0; k0 < DMODEL; k0 += 64) {
    __syncthreads();
    {  // A tile: bf16 verbatim -> async global->LDS, 4 x b128 per thread
      int row = tid >> 1;
      int kh  = (tid & 1) * 32;
      const __bf16* src = A + (size_t)(m0 + row) * DMODEL + k0 + kh;
      __bf16* dst = &ldsA[row * 64 + kh];
#pragma unroll
      for (int c = 0; c < 4; ++c) async_copy_b128(src + c * 8, dst + c * 8);
    }
    {  // B tile: Wo f32 -> bf16 transposed
      int k   = tid & 63;
      int n16 = (tid >> 6) * 16;
      const float* src = W + (size_t)(k0 + k) * DMODEL + n0 + n16;
#pragma unroll
      for (int i = 0; i < 16; ++i) ldsB[(n16 + i) * 64 + k] = (__bf16)src[i];
    }
    wait_async();
    __syncthreads();

    v16bf a0 = load_afrag(&ldsA[wave * 16 * 64], 64, lane, 0);
    v16bf a1 = load_afrag(&ldsA[wave * 16 * 64], 64, lane, 32);
#pragma unroll
    for (int nt = 0; nt < 4; ++nt) {
      acc[nt] = wmma_bf16(a0, load_bfrag(ldsB, 64, lane, nt, 0), acc[nt]);
      acc[nt] = wmma_bf16(a1, load_bfrag(ldsB, 64, lane, nt, 32), acc[nt]);
    }
  }

#pragma unroll
  for (int nt = 0; nt < 4; ++nt) {
    int d = nt * 16 + (lane & 15);
    float bv = bias[n0 + d];
#pragma unroll
    for (int r = 0; r < 8; ++r) {
      int row = m0 + wave * 16 + r + ((lane >> 4) << 3);
      Y[(size_t)row * DMODEL + n0 + d] = acc[nt][r] + bv;
    }
  }
}

// ---------------------------------------------------------------------------
extern "C" void kernel_launch(void* const* d_in, const int* in_sizes, int n_in,
                              void* d_out, int out_size, void* d_ws,
                              size_t ws_size, hipStream_t stream) {
  (void)in_sizes; (void)n_in; (void)out_size; (void)ws_size;
  const float* q    = (const float*)d_in[0];
  const float* v    = (const float*)d_in[1];
  const float* k    = (const float*)d_in[2];
  const int*   pos  = (const int*)d_in[3];
  const float* mask = (const float*)d_in[4];
  const float* Wq   = (const float*)d_in[5];
  const float* bq   = (const float*)d_in[6];
  const float* Wk   = (const float*)d_in[7];
  const float* bk   = (const float*)d_in[8];
  const float* Wv   = (const float*)d_in[9];
  const float* bv   = (const float*)d_in[10];
  const float* Wo   = (const float*)d_in[11];
  const float* bo   = (const float*)d_in[12];
  const float* rel  = (const float*)d_in[13];
  float* out = (float*)d_out;

  const size_t NPH = (size_t)NBATCH * NHEAD * S_LEN * DHEAD;  // per-tensor
  __bf16* ws  = (__bf16*)d_ws;
  __bf16* qh  = ws;
  __bf16* kh  = ws + NPH;
  __bf16* vh  = ws + 2 * NPH;
  __bf16* ao  = ws + 3 * NPH;  // merged-head attention output [B][S][768]

  dim3 gProj(32, 12, 1), bProj(256, 1, 1);
  proj_kernel<<<gProj, bProj, 0, stream>>>(q, Wq, bq, qh);
  proj_kernel<<<gProj, bProj, 0, stream>>>(k, Wk, bk, kh);
  proj_kernel<<<gProj, bProj, 0, stream>>>(v, Wv, bv, vh);

  dim3 gAttn(S_LEN / 64, NBATCH * NHEAD, 1), bAttn(128, 1, 1);
  attn_kernel<<<gAttn, bAttn, 0, stream>>>(qh, kh, vh, pos, mask, rel, ao);

  oproj_kernel<<<gProj, bProj, 0, stream>>>(ao, Wo, bo, out);
}
